// MultiHeadSpatialAttention_67070209295116
// MI455X (gfx1250) — matrix operand order repaired and checked by reading, not compiled, not run on hardware
//
#include <hip/hip_runtime.h>
#include <hip/hip_bf16.h>
#include <math.h>

// ---------------------------------------------------------------------------
// Types for CDNA5 WMMA (wave32): v_wmma_f32_16x16x32_bf16
// ---------------------------------------------------------------------------
typedef __bf16 bf16;
typedef bf16  v16bf __attribute__((ext_vector_type(16)));
typedef bf16  v8bf  __attribute__((ext_vector_type(8)));
typedef float v8f   __attribute__((ext_vector_type(8)));

#define NLAYER 4
#define CDIM   512
#define NTOK   1024
#define NHEADS 8
#define HDIM   64
#define BATCH  8
#define FDIM   2048   // 4*C
#define ATT_SCALE 0.125f  // HD^-0.5

__device__ __forceinline__ v8f wmma_bf16(v16bf a, v16bf b, v8f c) {
  return __builtin_amdgcn_wmma_f32_16x16x32_bf16(false, a, false, b, (short)0, c,
                                                 false, false);
}

// A fragment: 16x32 bf16, lane L holds row (L&15); element e is
// k = kbase + (L>=16 ? 16 : 0) + e.
__device__ __forceinline__ v16bf load_a_frag(const bf16* base, int lda, int lane) {
  const bf16* p = base + (size_t)(lane & 15) * lda + ((lane >> 4) & 1) * 16;
  v8bf lo = *(const v8bf*)p;
  v8bf hi = *(const v8bf*)(p + 8);
  v16bf r;
#pragma unroll
  for (int i = 0; i < 8; i++) { r[i] = lo[i]; r[i + 8] = hi[i]; }
  return r;
}

// B fragment: 32x16 (KxN) bf16 for NT gemm against row-major W(O,K):
// lane L -> column n = L&15 ; element e -> k = (L>=16?16:0)+e.
__device__ __forceinline__ v16bf load_b_frag(const bf16* base, int ldb, int lane) {
  const bf16* p = base + (size_t)(lane & 15) * ldb + ((lane >> 4) & 1) * 16;
  return *(const v16bf*)p;
}

// ---------------------------------------------------------------------------
// Elementwise helpers
// ---------------------------------------------------------------------------
__global__ void cvt_f32_bf16(const float* __restrict__ in, bf16* __restrict__ out, int n) {
  for (int i = blockIdx.x * blockDim.x + threadIdx.x; i < n; i += gridDim.x * blockDim.x)
    out[i] = (bf16)in[i];
}

// x: (B, C, N) f32  ->  h: (B, N, C) f32 (LDS-tiled transpose)
__global__ __launch_bounds__(256) void nchw_to_nlc(const float* __restrict__ x,
                                                   float* __restrict__ h) {
  __shared__ float tile[32][33];
  int b = blockIdx.z, n0 = blockIdx.x * 32, c0 = blockIdx.y * 32;
  for (int i = threadIdx.y; i < 32; i += 8)
    tile[i][threadIdx.x] = x[((size_t)b * CDIM + c0 + i) * NTOK + n0 + threadIdx.x];
  __syncthreads();
  for (int i = threadIdx.y; i < 32; i += 8)
    h[((size_t)b * NTOK + n0 + i) * CDIM + c0 + threadIdx.x] = tile[threadIdx.x][i];
}

// LayerNorm over C=512, f32 in -> bf16 out. One block (128 thr) per row.
__global__ __launch_bounds__(128) void layernorm_k(const float* __restrict__ h,
                                                   const float* __restrict__ w,
                                                   const float* __restrict__ b,
                                                   bf16* __restrict__ out) {
  int row = blockIdx.x;
  const float4* hr4 = (const float4*)(h + (size_t)row * CDIM);
  float4 vv = hr4[threadIdx.x];
  float s  = vv.x + vv.y + vv.z + vv.w;
  float s2 = vv.x * vv.x + vv.y * vv.y + vv.z * vv.z + vv.w * vv.w;
#pragma unroll
  for (int off = 16; off; off >>= 1) {
    s  += __shfl_xor(s, off, 32);
    s2 += __shfl_xor(s2, off, 32);
  }
  __shared__ float ps[4], ps2[4];
  int wv = threadIdx.x >> 5, ln = threadIdx.x & 31;
  if (ln == 0) { ps[wv] = s; ps2[wv] = s2; }
  __syncthreads();
  s = ps[0] + ps[1] + ps[2] + ps[3];
  s2 = ps2[0] + ps2[1] + ps2[2] + ps2[3];
  float mean = s * (1.f / CDIM);
  float var  = s2 * (1.f / CDIM) - mean * mean;
  float rstd = rsqrtf(var + 1e-5f);
  float vals[4] = {vv.x, vv.y, vv.z, vv.w};
#pragma unroll
  for (int i = 0; i < 4; i++) {
    int c = threadIdx.x * 4 + i;
    out[(size_t)row * CDIM + c] = (bf16)((vals[i] - mean) * rstd * w[c] + b[c]);
  }
}

// ---------------------------------------------------------------------------
// Generic NT GEMM: out[m,o] = epi( sum_k A[m,k]*W[o,k] )
// block = 4 waves; each wave owns a 32x64 output tile (2 A-frags x 4 B-frags,
// 8 WMMA per k-step). B-frags are preloaded so the compiler can clause the
// loads and hide them behind the WMMA burst; next k-tile is prefetched.
// ---------------------------------------------------------------------------
enum { EPI_BIAS_BF16 = 0, EPI_BIAS_VT = 1, EPI_BIAS_GELU_BF16 = 2,
       EPI_BIAS_RES_F32 = 3, EPI_SCALE_F32 = 4 };

template <int EPI>
__global__ __launch_bounds__(128) void gemm_nt(
    const bf16* __restrict__ A, int lda, long long sA,
    const bf16* __restrict__ W, int ldb, long long sB,
    const float* __restrict__ bias,
    void* __restrict__ out, int ldo, long long sO,
    const float* __restrict__ res,
    int K, float scale) {
  int lane = threadIdx.x & 31;
  int wave = threadIdx.x >> 5;
  int m0 = blockIdx.x * 128 + wave * 32;
  int o0 = blockIdx.y * 64;
  int bz = blockIdx.z;
  const bf16* Ab = A + (size_t)bz * sA + (size_t)m0 * lda;
  const bf16* Wb = W + (size_t)bz * sB;
  v8f acc[2][4] = {};
  for (int k0 = 0; k0 < K; k0 += 32) {
    v16bf a0 = load_a_frag(Ab + k0, lda, lane);
    v16bf a1 = load_a_frag(Ab + (size_t)16 * lda + k0, lda, lane);
    v16bf bfr[4];
#pragma unroll
    for (int j = 0; j < 4; j++)
      bfr[j] = load_b_frag(Wb + (size_t)(o0 + 16 * j) * ldb + k0, ldb, lane);
    if (k0 + 32 < K) {
      // pull next k-tile toward the caches while the WMMA burst runs
      __builtin_prefetch(Ab + (size_t)(lane & 15) * lda + k0 + 32, 0, 3);
      __builtin_prefetch(Ab + (size_t)(16 + (lane & 15)) * lda + k0 + 32, 0, 3);
      __builtin_prefetch(Wb + (size_t)(o0 + (lane & 31) * 2) * ldb + k0 + 32, 0, 3);
    }
#pragma unroll
    for (int j = 0; j < 4; j++) acc[0][j] = wmma_bf16(a0, bfr[j], acc[0][j]);
#pragma unroll
    for (int j = 0; j < 4; j++) acc[1][j] = wmma_bf16(a1, bfr[j], acc[1][j]);
  }
  int col = lane & 15, rbase = (lane >> 4) * 8;
#pragma unroll
  for (int r = 0; r < 2; r++) {
#pragma unroll
    for (int j = 0; j < 4; j++) {
#pragma unroll
      for (int v = 0; v < 8; v++) {
        int m = m0 + 16 * r + rbase + v;
        int o = o0 + 16 * j + col;
        float xv = acc[r][j][v];
        if constexpr (EPI == EPI_SCALE_F32) {
          __builtin_nontemporal_store(
              xv * scale, &((float*)out)[(size_t)bz * sO + (size_t)m * ldo + o]);
        } else if constexpr (EPI == EPI_BIAS_RES_F32) {
          xv += bias[o];
          ((float*)out)[(size_t)m * ldo + o] = res[(size_t)m * ldo + o] + xv;
        } else if constexpr (EPI == EPI_BIAS_GELU_BF16) {
          xv += bias[o];
          xv = 0.5f * xv * (1.f + erff(xv * 0.70710678118f));
          ((bf16*)out)[(size_t)m * ldo + o] = (bf16)xv;
        } else if constexpr (EPI == EPI_BIAS_VT) {
          xv += bias[o];
          int bb = m >> 10, n = m & (NTOK - 1);
          int hh = o >> 6, dd = o & (HDIM - 1);
          // vT layout: (B, H, HD, N)
          ((bf16*)out)[(((size_t)(bb * NHEADS + hh)) * HDIM + dd) * NTOK + n] = (bf16)xv;
        } else {  // EPI_BIAS_BF16
          xv += bias[o];
          ((bf16*)out)[(size_t)m * ldo + o] = (bf16)xv;
        }
      }
    }
  }
}

// ---------------------------------------------------------------------------
// Fused flash attention (layers 0..2): softmax(q k^T * s) v  -> ao (B*N, C)
// One wave per 16 query rows of one (b, head). Online softmax via LDS.
// q,k: (B*N, C) bf16 ; vt: (B, H, HD, N) bf16 ; ao: (B*N, C) bf16
// ---------------------------------------------------------------------------
__global__ __launch_bounds__(32) void flash_attn(const bf16* __restrict__ q,
                                                 const bf16* __restrict__ k,
                                                 const bf16* __restrict__ vt,
                                                 bf16* __restrict__ ao) {
  __shared__ __align__(16) float sS[16 * 32];
  __shared__ __align__(16) bf16  sP[16 * 32];
  __shared__ float sCorr[16];
  int lane = threadIdx.x;
  int n0 = blockIdx.x * 16, h = blockIdx.y, b = blockIdx.z;
  const bf16* qb = q + ((size_t)(b * NTOK + n0)) * CDIM + h * HDIM;
  const bf16* kb = k + ((size_t)b * NTOK) * CDIM + h * HDIM;
  const bf16* vb = vt + ((size_t)(b * NHEADS + h)) * HDIM * NTOK;

  v16bf qa0 = load_a_frag(qb, CDIM, lane);        // head dims 0..31
  v16bf qa1 = load_a_frag(qb + 32, CDIM, lane);   // head dims 32..63

  int col = lane & 15, rbase = (lane >> 4) * 8;
  float rmax = -1e30f, rsum = 0.f;                // valid in lanes 0..15
  v8f acc[4] = {};

  for (int m0 = 0; m0 < NTOK; m0 += 32) {
    v16bf kf0  = load_b_frag(kb + (size_t)m0 * CDIM, CDIM, lane);
    v16bf kf1  = load_b_frag(kb + (size_t)m0 * CDIM + 32, CDIM, lane);
    v16bf kf2  = load_b_frag(kb + (size_t)(m0 + 16) * CDIM, CDIM, lane);
    v16bf kf3  = load_b_frag(kb + (size_t)(m0 + 16) * CDIM + 32, CDIM, lane);
    v8f s0 = {}, s1 = {};
    s0 = wmma_bf16(qa0, kf0, s0);
    s0 = wmma_bf16(qa1, kf1, s0);
    s1 = wmma_bf16(qa0, kf2, s1);
    s1 = wmma_bf16(qa1, kf3, s1);
#pragma unroll
    for (int v = 0; v < 8; v++) {
      int r = rbase + v;
      sS[r * 32 + col]      = s0[v];
      sS[r * 32 + 16 + col] = s1[v];
    }
    __syncthreads();
    if (lane < 16) {
      float vmax = -1e30f;
#pragma unroll 8
      for (int j = 0; j < 32; j++) vmax = fmaxf(vmax, sS[lane * 32 + j] * ATT_SCALE);
      float newmax = fmaxf(rmax, vmax);
      float corr = __expf(rmax - newmax);
      float ssum = 0.f;
#pragma unroll 8
      for (int j = 0; j < 32; j++) {
        float p = __expf(sS[lane * 32 + j] * ATT_SCALE - newmax);
        sP[lane * 32 + j] = (bf16)p;
        ssum += p;
      }
      rsum = rsum * corr + ssum;
      rmax = newmax;
      sCorr[lane] = corr;
    }
    __syncthreads();
    v16bf pa = load_a_frag(sP, 32, lane);
    v16bf vf[4];
#pragma unroll
    for (int j = 0; j < 4; j++)
      vf[j] = load_b_frag(vb + (size_t)(j * 16) * NTOK + m0, NTOK, lane);
#pragma unroll
    for (int j = 0; j < 4; j++) {
#pragma unroll
      for (int v = 0; v < 8; v++) acc[j][v] *= sCorr[rbase + v];
      acc[j] = wmma_bf16(pa, vf[j], acc[j]);
    }
    __syncthreads();
  }
  if (lane < 16) sCorr[lane] = 1.0f / rsum;
  __syncthreads();
  bf16* aob = ao + ((size_t)(b * NTOK + n0)) * CDIM + h * HDIM;
#pragma unroll
  for (int j = 0; j < 4; j++)
#pragma unroll
    for (int v = 0; v < 8; v++)
      aob[(size_t)(rbase + v) * CDIM + j * 16 + col] = (bf16)(acc[j][v] * sCorr[rbase + v]);
}

// ---------------------------------------------------------------------------
// Host orchestration
// ---------------------------------------------------------------------------
extern "C" void kernel_launch(void* const* d_in, const int* in_sizes, int n_in,
                              void* d_out, int out_size, void* d_ws, size_t ws_size,
                              hipStream_t stream) {
  const float* x    = (const float*)d_in[0];
  const float* Wq   = (const float*)d_in[1];  const float* bq  = (const float*)d_in[2];
  const float* Wk   = (const float*)d_in[3];  const float* bk  = (const float*)d_in[4];
  const float* Wv   = (const float*)d_in[5];  const float* bv  = (const float*)d_in[6];
  const float* Wo   = (const float*)d_in[7];  const float* bo  = (const float*)d_in[8];
  const float* ln1w = (const float*)d_in[9];  const float* ln1b = (const float*)d_in[10];
  const float* ln2w = (const float*)d_in[11]; const float* ln2b = (const float*)d_in[12];
  const float* Wf1  = (const float*)d_in[13]; const float* bf1 = (const float*)d_in[14];
  const float* Wf2  = (const float*)d_in[15]; const float* bf2 = (const float*)d_in[16];

  char* cur = (char*)d_ws;
  auto alloc = [&](size_t bytes) -> void* {
    void* r = cur;
    cur += (bytes + 255) & ~(size_t)255;
    return r;
  };
  const size_t CC = (size_t)CDIM * CDIM;      // 262144
  const size_t FC = (size_t)FDIM * CDIM;      // 1048576
  const size_t MN = (size_t)BATCH * NTOK;     // 8192 rows

  bf16* wq_b  = (bf16*)alloc(NLAYER * CC * 2);
  bf16* wk_b  = (bf16*)alloc(NLAYER * CC * 2);
  bf16* wv_b  = (bf16*)alloc(NLAYER * CC * 2);
  bf16* wo_b  = (bf16*)alloc(NLAYER * CC * 2);
  bf16* wf1_b = (bf16*)alloc(NLAYER * FC * 2);
  bf16* wf2_b = (bf16*)alloc(NLAYER * FC * 2);
  float* h    = (float*)alloc(MN * CDIM * 4);
  bf16* hn    = (bf16*)alloc(MN * CDIM * 2);
  bf16* qb    = (bf16*)alloc(MN * CDIM * 2);
  bf16* kb    = (bf16*)alloc(MN * CDIM * 2);
  bf16* vt    = (bf16*)alloc(MN * CDIM * 2);
  bf16* ao    = (bf16*)alloc(MN * CDIM * 2);
  bf16* f1    = (bf16*)alloc(MN * FDIM * 2);

  // Weight conversion to bf16 (L2-resident thereafter).
  cvt_f32_bf16<<<2048, 256, 0, stream>>>(Wq,  wq_b,  (int)(NLAYER * CC));
  cvt_f32_bf16<<<2048, 256, 0, stream>>>(Wk,  wk_b,  (int)(NLAYER * CC));
  cvt_f32_bf16<<<2048, 256, 0, stream>>>(Wv,  wv_b,  (int)(NLAYER * CC));
  cvt_f32_bf16<<<2048, 256, 0, stream>>>(Wo,  wo_b,  (int)(NLAYER * CC));
  cvt_f32_bf16<<<2048, 256, 0, stream>>>(Wf1, wf1_b, (int)(NLAYER * FC));
  cvt_f32_bf16<<<2048, 256, 0, stream>>>(Wf2, wf2_b, (int)(NLAYER * FC));

  // (B,C,H,W) -> (B,N,C)
  nchw_to_nlc<<<dim3(NTOK / 32, CDIM / 32, BATCH), dim3(32, 8), 0, stream>>>(x, h);

  dim3 blk(128);
  dim3 gProj(MN / 128, CDIM / 64, 1);   // (64, 8)
  dim3 gF1(MN / 128, FDIM / 64, 1);     // (64, 32)

  for (int l = 0; l < NLAYER; l++) {
    layernorm_k<<<(int)MN, 128, 0, stream>>>(h, ln1w + l * CDIM, ln1b + l * CDIM, hn);
    gemm_nt<EPI_BIAS_BF16><<<gProj, blk, 0, stream>>>(
        hn, CDIM, 0, wq_b + l * CC, CDIM, 0, bq + l * CDIM, qb, CDIM, 0, nullptr,
        CDIM, 0.f);
    gemm_nt<EPI_BIAS_BF16><<<gProj, blk, 0, stream>>>(
        hn, CDIM, 0, wk_b + l * CC, CDIM, 0, bk + l * CDIM, kb, CDIM, 0, nullptr,
        CDIM, 0.f);
    if (l < NLAYER - 1) {
      gemm_nt<EPI_BIAS_VT><<<gProj, blk, 0, stream>>>(
          hn, CDIM, 0, wv_b + l * CC, CDIM, 0, bv + l * CDIM, vt, CDIM, 0, nullptr,
          CDIM, 0.f);
      flash_attn<<<dim3(NTOK / 16, NHEADS, BATCH), 32, 0, stream>>>(qb, kb, vt, ao);
      gemm_nt<EPI_BIAS_RES_F32><<<gProj, blk, 0, stream>>>(
          ao, CDIM, 0, wo_b + l * CC, CDIM, 0, bo + l * CDIM, h, CDIM, 0, h,
          CDIM, 0.f);
      layernorm_k<<<(int)MN, 128, 0, stream>>>(h, ln2w + l * CDIM, ln2b + l * CDIM, hn);
      gemm_nt<EPI_BIAS_GELU_BF16><<<gF1, blk, 0, stream>>>(
          hn, CDIM, 0, wf1_b + l * FC, CDIM, 0, bf1 + l * FDIM, f1, FDIM, 0, nullptr,
          CDIM, 0.f);
      gemm_nt<EPI_BIAS_RES_F32><<<gProj, blk, 0, stream>>>(
          f1, FDIM, 0, wf2_b + l * FC, FDIM, 0, bf2 + l * CDIM, h, CDIM, 0, h,
          FDIM, 0.f);
    } else {
      // attn_weights = (scale/HEADS) * q3 @ k3^T over full C (batched over B)
      gemm_nt<EPI_SCALE_F32><<<dim3(NTOK / 128, NTOK / 64, BATCH), blk, 0, stream>>>(
          qb, CDIM, (long long)NTOK * CDIM, kb, CDIM, (long long)NTOK * CDIM, nullptr,
          d_out, NTOK, (long long)NTOK * NTOK, nullptr, CDIM,
          ATT_SCALE / (float)NHEADS);
    }
  }
  (void)in_sizes; (void)n_in; (void)out_size; (void)ws_size;
}
